// Attention_77309411430
// MI455X (gfx1250) — matrix operand order repaired and checked by reading, not compiled, not run on hardware
//
#include <hip/hip_runtime.h>
#include <hip/hip_bf16.h>

// ---------------------------------------------------------------------------
// Fused MHA for MI455X (gfx1250, wave32, WMMA + TDM).
//   B=2, S=4096, D=512, H=8, HD=64
// Pass 1: QKV projections via v_wmma_f32_16x16x32_bf16 (bias fused, Q
//         pre-scaled by 1/sqrt(HD)); bf16 Q,K as [B*H, S, 64], V transposed
//         [B*H, 64, S].
// Pass 2: flash attention. K/V chunks staged into double-buffered LDS by the
//         Tensor Data Mover (tensor_load_to_lds, TENSORcnt-synchronized),
//         shared by 8 waves/block; each wave owns 16 query rows with online
//         softmax; QK^T and P*V via v_wmma_f32_16x16x32_bf16.
// ---------------------------------------------------------------------------

typedef __attribute__((ext_vector_type(16))) __bf16 v16bf;
typedef __attribute__((ext_vector_type(8)))  __bf16 v8bf;
typedef __attribute__((ext_vector_type(8)))  float  v8f;
typedef __attribute__((ext_vector_type(4)))  unsigned int u32x4;
typedef __attribute__((ext_vector_type(8)))  int i32x8;
typedef __attribute__((ext_vector_type(4)))  int i32x4;

#define B_  2
#define S_  4096
#define D_  512
#define H_  8
#define HD_ 64

#define KPAD 72   // kbuf row stride (halves): 64 data + 8 pad = 144B
#define VPAD 40   // vbuf row stride (halves): 32 data + 8 pad =  80B

#if defined(__has_builtin)
#if __has_builtin(__builtin_amdgcn_wave_barrier)
#define WAVE_FENCE() __builtin_amdgcn_wave_barrier()
#endif
#if __has_builtin(__builtin_amdgcn_tensor_load_to_lds) && \
    __has_builtin(__builtin_amdgcn_s_wait_tensorcnt)
#define HAS_TDM 1
#endif
#endif
#ifndef WAVE_FENCE
#define WAVE_FENCE() asm volatile("" ::: "memory")
#endif

// 16 contiguous bf16 -> B-fragment vector (two aligned 16B loads)
__device__ __forceinline__ v16bf load16(const __bf16* p) {
    v8bf lo = *(const v8bf*)(p);
    v8bf hi = *(const v8bf*)(p + 8);
    v16bf r;
    #pragma unroll
    for (int j = 0; j < 8; ++j) { r[j] = lo[j]; r[8 + j] = hi[j]; }
    return r;
}

#ifdef HAS_TDM
// Issue one 2D TDM load: global (row-major, elem=2B) -> padded LDS tile.
// Descriptor packing per CDNA5 ISA §8.3/8.4 (D# groups 0/1; groups 2/3 zero
// for 2D tiles). pad_interval: rows of 2^(pint+1) DWORDs; pad_amount: pamt+1
// DWORDs appended per row. clang-23 builtin takes 6 args (extra zero v8i32).
__device__ __forceinline__ void tdm_load_2d(unsigned lds_addr, const void* gptr,
                                            unsigned tensor_d0, unsigned tensor_d1,
                                            unsigned tile_d0, unsigned tile_d1,
                                            unsigned stride0,
                                            unsigned pint, unsigned pamt)
{
    const unsigned long long ga = (unsigned long long)gptr;
    u32x4 g0;
    g0[0] = 1u;                                           // count=1 (valid, user)
    g0[1] = lds_addr;                                     // LDS byte address
    g0[2] = (unsigned)(ga & 0xFFFFFFFFull);               // global_addr[31:0]
    g0[3] = (unsigned)((ga >> 32) & 0x01FFFFFFull)        // global_addr[56:32]
          | 0x80000000u;                                  // type=2 ("image")
    i32x8 g1;
    g1[0] = (int)((1u << 16)            // data_size = 1 -> 2 bytes
          |       (1u << 20)            // pad_enable
          |       (pint << 22) | (pamt << 25));
    g1[1] = (int)((tensor_d0 & 0xFFFFu) << 16);           // tensor_dim0[15:0]
    g1[2] = (int)(((tensor_d0 >> 16) & 0xFFFFu)
          |       ((tensor_d1 & 0xFFFFu) << 16));         // dim0[31:16] | dim1[15:0]
    g1[3] = (int)(((tensor_d1 >> 16) & 0xFFFFu)
          |       (tile_d0 << 16));                       // dim1[31:16] | tile_dim0
    g1[4] = (int)(tile_d1 & 0xFFFFu);                     // tile_dim1 (tile_dim2=0)
    g1[5] = (int)stride0;                                 // tensor_dim0_stride[31:0]
    g1[6] = 0;                                            // stride0[47:32] | stride1 lo
    g1[7] = 0;
    const i32x4 z4 = {0, 0, 0, 0};
    const i32x8 z8 = {0, 0, 0, 0, 0, 0, 0, 0};
    __builtin_amdgcn_tensor_load_to_lds(g0, g1, z4, z4, z8, 0);
}
#endif

// ---------------------------------------------------------------------------
// Pass 1: out = (X @ W + b) * outScale   (X:[B*S,512] f32, W:[512,512] f32)
// One wave computes a 16x16 output tile; 4 waves per block along N.
// transposeV=0: store [(b*H+h)*S + s]*64 + d      (Q, K layout)
// transposeV=1: store [(b*H+h)*64 + d]*S + s      (V^T layout for P*V B-frags)
// ---------------------------------------------------------------------------
__global__ __launch_bounds__(128)
void qkv_proj_kernel(const float* __restrict__ X, const float* __restrict__ W,
                     const float* __restrict__ bias, __bf16* __restrict__ out,
                     int transposeV, float outScale)
{
    const int lane = threadIdx.x & 31;
    const int wave = threadIdx.x >> 5;
    const int hi   = lane >> 4;
    const int l15  = lane & 15;
    const int m0   = blockIdx.x * 16;                  // row tile in [0, B*S)
    const int n0   = (blockIdx.y * 4 + wave) * 16;     // col tile in [0, D)

    v8f acc = {};

    for (int kc = 0; kc < D_; kc += 32) {
        // --- A fragment (16x32 bf16): lane = row M, K = (j<8?j:j+8)+8*hi ---
        const float* xrow = X + (size_t)(m0 + l15) * D_ + kc + hi * 8;
        float4 a0 = *(const float4*)(xrow);        // K = base+0..3
        float4 a1 = *(const float4*)(xrow + 4);    // K = base+4..7
        float4 a2 = *(const float4*)(xrow + 16);   // K = base+16..19
        float4 a3 = *(const float4*)(xrow + 20);   // K = base+20..23
        v16bf af;
        af[0]=(__bf16)a0.x;  af[1]=(__bf16)a0.y;  af[2]=(__bf16)a0.z;  af[3]=(__bf16)a0.w;
        af[4]=(__bf16)a1.x;  af[5]=(__bf16)a1.y;  af[6]=(__bf16)a1.z;  af[7]=(__bf16)a1.w;
        af[8]=(__bf16)a2.x;  af[9]=(__bf16)a2.y;  af[10]=(__bf16)a2.z; af[11]=(__bf16)a2.w;
        af[12]=(__bf16)a3.x; af[13]=(__bf16)a3.y; af[14]=(__bf16)a3.z; af[15]=(__bf16)a3.w;

        // --- B fragment (32x16 bf16): lane = col N, K = (j<8?j:j+8)+8*hi ---
        const float* wcol = W + (size_t)(kc + hi * 8) * D_ + n0 + l15;
        v16bf bfrag;
        #pragma unroll
        for (int j = 0; j < 16; ++j)
            bfrag[j] = (__bf16)wcol[(size_t)(j < 8 ? j : j + 8) * D_];

        acc = __builtin_amdgcn_wmma_f32_16x16x32_bf16(
            false, af, false, bfrag, (short)0, acc, false, false);
    }

    // --- bias + scale + store ----------------------------------------------
    const int   col = n0 + l15;
    const float bv  = bias[col];
    const int   h   = col >> 6;
    const int   d   = col & 63;
    #pragma unroll
    for (int r = 0; r < 8; ++r) {
        const int row  = m0 + r + hi * 8;          // row in [0, B*S)
        const int bidx = row >> 12;                // / S_
        const int s    = row & (S_ - 1);
        const float v  = (acc[r] + bv) * outScale;
        size_t idx;
        if (transposeV) idx = ((size_t)(bidx * H_ + h) * HD_ + d) * S_ + s;
        else            idx = ((size_t)(bidx * H_ + h) * S_  + s) * HD_ + d;
        out[idx] = (__bf16)v;
    }
}

// ---------------------------------------------------------------------------
// Pass 2: flash attention, TDM double-buffered K/V staging in LDS.
// ---------------------------------------------------------------------------
__global__ __launch_bounds__(256)
void flash_attn_kernel(const __bf16* __restrict__ Qb, const __bf16* __restrict__ Kb,
                       const __bf16* __restrict__ Vt, float* __restrict__ out)
{
    __shared__ __bf16 kbuf[2][32][KPAD];   // 32 keys x 64 d (padded), x2   ~9.2 KB
    __shared__ __bf16 vbuf[2][64][VPAD];   // 64 d    x 32 keys (padded), x2 ~10.2 KB
    __shared__ __bf16 pbuf[8][16][32];     // per-wave P staging               8 KB

    const int tid  = threadIdx.x;
    const int lane = tid & 31;
    const int wave = tid >> 5;
    const int hi   = lane >> 4;
    const int l15  = lane & 15;
    const int bh   = blockIdx.y;                       // 0 .. B*H-1
    const int q0   = blockIdx.x * 128 + wave * 16;     // query row tile

    const __bf16* Qbh = Qb + (size_t)bh * S_ * HD_;
    const __bf16* Kbh = Kb + (size_t)bh * S_ * HD_;
    const __bf16* Vbh = Vt + (size_t)bh * HD_ * S_;

    const int NCH = S_ / 32;   // 128 key chunks

#ifdef HAS_TDM
    // prologue: chunks 0 and 1 in flight via the Tensor Data Mover
    if (wave == 0) {
        tdm_load_2d((unsigned)(uintptr_t)&kbuf[0][0][0], Kbh,
                    HD_, S_, HD_, 32, HD_, /*pint=*/4, /*pamt=*/3);
        tdm_load_2d((unsigned)(uintptr_t)&vbuf[0][0][0], Vbh,
                    S_, HD_, 32, HD_, S_, /*pint=*/3, /*pamt=*/3);
        tdm_load_2d((unsigned)(uintptr_t)&kbuf[1][0][0], Kbh + (size_t)32 * HD_,
                    HD_, S_, HD_, 32, HD_, 4, 3);
        tdm_load_2d((unsigned)(uintptr_t)&vbuf[1][0][0], Vbh + 32,
                    S_, HD_, 32, HD_, S_, 3, 3);
        __builtin_amdgcn_s_wait_tensorcnt(2);   // chunk 0 resident (in-order)
    }
    __syncthreads();
#else
    // fallback: cooperative register-staged copies (256 threads, one uint4 each)
    const int krow_s = tid >> 3, kcol_s = (tid & 7) * 8;   // K chunk: 32 x 64
    const int vrow_s = tid >> 2, vcol_s = (tid & 3) * 8;   // V chunk: 64 x 32
    uint4 kreg = *(const uint4*)(Kbh + (size_t)krow_s * HD_ + kcol_s);
    uint4 vreg = *(const uint4*)(Vbh + (size_t)vrow_s * S_ + vcol_s);
#endif

    // --- load Q A-fragments, K-dim chunks d=[0,32) and [32,64) -------------
    v16bf qf[2];
    {
        const __bf16* qrow = Qbh + (size_t)(q0 + l15) * HD_;
        #pragma unroll
        for (int c = 0; c < 2; ++c) {
            const __bf16* p = qrow + c * 32 + hi * 16;
            v8bf lo  = *(const v8bf*)(p);        // K = base + 0..7
            v8bf hic = *(const v8bf*)(p + 8);    // K = base + 8..15
            #pragma unroll
            for (int j = 0; j < 8; ++j) { qf[c][j] = lo[j]; qf[c][8 + j] = hic[j]; }
        }
    }

    v8f acc[4] = {v8f{}, v8f{}, v8f{}, v8f{}};   // 16 rows x 64 d
    float mrow[8], lrow[8];
    #pragma unroll
    for (int r = 0; r < 8; ++r) { mrow[r] = -1e30f; lrow[r] = 0.0f; }

    for (int i = 0; i < NCH; ++i) {
        const int cur = i & 1;

#ifndef HAS_TDM
        *(uint4*)&kbuf[cur][krow_s][kcol_s] = kreg;
        *(uint4*)&vbuf[cur][vrow_s][vcol_s] = vreg;
        __syncthreads();
        if (i + 1 < NCH) {
            const int nk = (i + 1) * 32;
            kreg = *(const uint4*)(Kbh + (size_t)(nk + krow_s) * HD_ + kcol_s);
            vreg = *(const uint4*)(Vbh + (size_t)vrow_s * S_ + nk + vcol_s);
        }
#endif

        // --- scores: two 16x16 tiles, K-dim = 64 (Q pre-scaled) ------------
        v8f st[2] = {v8f{}, v8f{}};
        #pragma unroll
        for (int kk = 0; kk < 2; ++kk) {
            const __bf16* kr = &kbuf[cur][kk * 16 + l15][hi * 16];
            v16bf kf0 = load16(kr);        // d = 16*hi + 0..15
            v16bf kf1 = load16(kr + 32);   // d = 32 + 16*hi + 0..15
            st[kk] = __builtin_amdgcn_wmma_f32_16x16x32_bf16(
                false, qf[0], false, kf0, (short)0, st[kk], false, false);
            st[kk] = __builtin_amdgcn_wmma_f32_16x16x32_bf16(
                false, qf[1], false, kf1, (short)0, st[kk], false, false);
        }

        // --- online softmax ------------------------------------------------
        #pragma unroll
        for (int r = 0; r < 8; ++r) {
            float v = fmaxf(st[0][r], st[1][r]);
            v = fmaxf(v, __shfl_xor(v, 1, 32));
            v = fmaxf(v, __shfl_xor(v, 2, 32));
            v = fmaxf(v, __shfl_xor(v, 4, 32));
            v = fmaxf(v, __shfl_xor(v, 8, 32));
            const float mn    = fmaxf(mrow[r], v);
            const float alpha = __expf(mrow[r] - mn);
            mrow[r] = mn;
            const float p0 = __expf(st[0][r] - mn);
            const float p1 = __expf(st[1][r] - mn);
            st[0][r] = p0; st[1][r] = p1;
            float rs = p0 + p1;
            rs += __shfl_xor(rs, 1, 32);
            rs += __shfl_xor(rs, 2, 32);
            rs += __shfl_xor(rs, 4, 32);
            rs += __shfl_xor(rs, 8, 32);
            lrow[r] = lrow[r] * alpha + rs;
            #pragma unroll
            for (int t = 0; t < 4; ++t) acc[t][r] *= alpha;
        }

        // --- P: C-layout -> row-major LDS -> A-fragment (same-wave, in-order)
        __bf16* pl = &pbuf[wave][0][0];
        #pragma unroll
        for (int r = 0; r < 8; ++r) {
            pl[(r + hi * 8) * 32 + l15]      = (__bf16)st[0][r];
            pl[(r + hi * 8) * 32 + 16 + l15] = (__bf16)st[1][r];
        }
        WAVE_FENCE();
        v16bf pf;
        {
            const __bf16* pr = pl + l15 * 32 + hi * 16;
            v8bf lo  = *(const v8bf*)(pr);
            v8bf hic = *(const v8bf*)(pr + 8);
            #pragma unroll
            for (int j = 0; j < 8; ++j) { pf[j] = lo[j]; pf[8 + j] = hic[j]; }
        }

        // --- acc += P(16x32) @ V(32x16) per d-tile -------------------------
        #pragma unroll
        for (int t = 0; t < 4; ++t) {
            v16bf vf = load16(&vbuf[cur][t * 16 + l15][hi * 16]);  // k = 16*hi + j
            acc[t] = __builtin_amdgcn_wmma_f32_16x16x32_bf16(
                false, pf, false, vf, (short)0, acc[t], false, false);
        }

        __syncthreads();   // all waves done reading buffer `cur`

#ifdef HAS_TDM
        if (wave == 0) {
            if (i + 2 < NCH) {
                const int nk = (i + 2) * 32;
                tdm_load_2d((unsigned)(uintptr_t)&kbuf[cur][0][0],
                            Kbh + (size_t)nk * HD_, HD_, S_, HD_, 32, HD_, 4, 3);
                tdm_load_2d((unsigned)(uintptr_t)&vbuf[cur][0][0],
                            Vbh + nk, S_, HD_, 32, HD_, S_, 3, 3);
                __builtin_amdgcn_s_wait_tensorcnt(2);   // chunk i+1 resident
            } else {
                __builtin_amdgcn_s_wait_tensorcnt(0);   // drain tail
            }
        }
        __syncthreads();   // release waves onto buffer cur^1
#endif
    }

    // --- normalize and write [B,S,D] f32 -----------------------------------
    const int bidx = bh >> 3;
    const int h    = bh & 7;
    #pragma unroll
    for (int r = 0; r < 8; ++r) {
        const float inv  = 1.0f / lrow[r];
        const int   srow = q0 + r + hi * 8;
        float* orow = out + ((size_t)bidx * S_ + srow) * D_ + h * HD_;
        #pragma unroll
        for (int t = 0; t < 4; ++t)
            orow[t * 16 + l15] = acc[t][r] * inv;
    }
}

// ---------------------------------------------------------------------------
extern "C" void kernel_launch(void* const* d_in, const int* in_sizes, int n_in,
                              void* d_out, int out_size, void* d_ws, size_t ws_size,
                              hipStream_t stream)
{
    const float* query = (const float*)d_in[0];
    const float* key   = (const float*)d_in[1];
    const float* value = (const float*)d_in[2];
    const float* Wq    = (const float*)d_in[3];
    const float* bq    = (const float*)d_in[4];
    const float* Wk    = (const float*)d_in[5];
    const float* bk    = (const float*)d_in[6];
    const float* Wv    = (const float*)d_in[7];
    const float* bv    = (const float*)d_in[8];
    float* out = (float*)d_out;

    const size_t elems = (size_t)B_ * H_ * S_ * HD_;   // 4 Mi elements
    __bf16* Qb = (__bf16*)d_ws;
    __bf16* Kb = Qb + elems;
    __bf16* Vt = Kb + elems;                           // 24 MB total bf16

    const float qscale = 0.125f;                       // 1/sqrt(HD)

    dim3 pblk(128, 1, 1);
    dim3 pgrd((B_ * S_) / 16, D_ / 64, 1);             // 512 x 8 blocks
    qkv_proj_kernel<<<pgrd, pblk, 0, stream>>>(query, Wq, bq, Qb, 0, qscale);
    qkv_proj_kernel<<<pgrd, pblk, 0, stream>>>(key,   Wk, bk, Kb, 0, 1.0f);
    qkv_proj_kernel<<<pgrd, pblk, 0, stream>>>(value, Wv, bv, Vt, 1, 1.0f);

    dim3 ablk(256, 1, 1);
    dim3 agrd(S_ / 128, B_ * H_, 1);                   // 32 x 16 blocks
    flash_attn_kernel<<<agrd, ablk, 0, stream>>>(Qb, Kb, Vt, out);
}
